// HLFE_12120397709720
// MI455X (gfx1250) — matrix operand order repaired and checked by reading, not compiled, not run on hardware
//
#include <hip/hip_runtime.h>
#include <hip/hip_bf16.h>

// ---------------------------------------------------------------------------
// MI455X (gfx1250) HLFE forward.
// bf16 WMMA (v_wmma_f32_16x16x32_bf16) for all GEMMs, f32 accumulate.
// Activations kept pixel-major [P][C] so GEMM B-tiles stage as straight
// vector copies (no transpose in the hot loop).
// ---------------------------------------------------------------------------

typedef __bf16 bf16_t;
typedef __attribute__((ext_vector_type(8)))  __bf16 bf16x8;
typedef __attribute__((ext_vector_type(16))) __bf16 v16bf;
typedef __attribute__((ext_vector_type(8)))  float  v8f;

union Frag16 { v16bf v; bf16x8 h8[2]; };

#define APITCH 40   // halves per LDS row (K=32 + pad, 80B rows -> 16B aligned)
#define BPITCH 40

__device__ __forceinline__ bf16_t f2bf(float f)  { return (bf16_t)f; }
__device__ __forceinline__ float  bf2f(bf16_t b) { return (float)b; }

// ---------------------------------------------------------------------------
// f32 -> bf16 pack (weights, kept [O][C] row-major)
// ---------------------------------------------------------------------------
__global__ void k_f32_to_bf16(const float* __restrict__ in,
                              bf16_t* __restrict__ out, int n)
{
    int i = blockIdx.x * 256 + threadIdx.x;
    if (i < n) out[i] = f2bf(in[i]);
}

// ---------------------------------------------------------------------------
// x (B,512,4096) f32 -> x_t (B,4096,512) bf16  (LDS tile transpose)
// block handles 32 channels x 64 pixels; grid (64, 16, B)
// ---------------------------------------------------------------------------
__global__ void k_transpose_x(const float* __restrict__ x,
                              bf16_t* __restrict__ xt)
{
    __shared__ bf16_t tile[32 * 72];       // [c][p], pitch 72 (144B, 16B-aligned)
    const int pb = blockIdx.x * 64;
    const int cb = blockIdx.y * 32;
    const int b  = blockIdx.z;
    const int t  = threadIdx.x;
    {
        int c  = t >> 3;                   // 0..31
        int px = (t & 7) * 8;              // 0..56
        const float* src = x + (((size_t)(b * 512 + cb + c)) << 12) + pb + px;
        float4 f0 = *(const float4*)(src);
        float4 f1 = *(const float4*)(src + 4);
        bf16x8 v;
        v[0] = f2bf(f0.x); v[1] = f2bf(f0.y); v[2] = f2bf(f0.z); v[3] = f2bf(f0.w);
        v[4] = f2bf(f1.x); v[5] = f2bf(f1.y); v[6] = f2bf(f1.z); v[7] = f2bf(f1.w);
        *(bf16x8*)&tile[c * 72 + px] = v;
    }
    __syncthreads();
    {
        int p  = t >> 2;                   // 0..63
        int c8 = (t & 3) * 8;              // 0..24
        bf16x8 v;
#pragma unroll
        for (int j = 0; j < 8; ++j) v[j] = tile[(c8 + j) * 72 + p];
        *(bf16x8*)(xt + (((size_t)b) << 21) + (size_t)(pb + p) * 512 + cb + c8) = v;
    }
}

// ---------------------------------------------------------------------------
// 2x2 avg pool on pixel-major layout: xp_t[b][oy*32+ox][c] from x_t
// grid (1024, B), 256 threads (2 channels each) -- fully coalesced
// ---------------------------------------------------------------------------
__global__ void k_pool_t(const bf16_t* __restrict__ xt,
                         bf16_t* __restrict__ xpt)
{
    int pp = blockIdx.x;                   // 0..1023
    int b  = blockIdx.y;
    int oy = pp >> 5, ox = pp & 31;
    size_t base = (((size_t)b) << 21) + ((size_t)((oy * 2) * 64 + ox * 2) << 9);
    const bf16_t* r0 = xt + base;          // pixel (2oy, 2ox)
    const bf16_t* r1 = r0 + 512;           // (2oy, 2ox+1)
    const bf16_t* r2 = r0 + 64 * 512;      // (2oy+1, 2ox)
    const bf16_t* r3 = r2 + 512;           // (2oy+1, 2ox+1)
    int c = threadIdx.x * 2;
    float a0 = (bf2f(r0[c])   + bf2f(r1[c])   + bf2f(r2[c])   + bf2f(r3[c]))   * 0.25f;
    float a1 = (bf2f(r0[c+1]) + bf2f(r1[c+1]) + bf2f(r2[c+1]) + bf2f(r3[c+1])) * 0.25f;
    bf16_t* dst = xpt + (((size_t)b) << 19) + (size_t)pp * 512 + c;
    dst[0] = f2bf(a0);
    dst[1] = f2bf(a1);
}

// ---------------------------------------------------------------------------
// WMMA GEMM: Out[b] = A (OxC bf16 row-major) * B_t[b] (PxC bf16, K contiguous)
// 256 threads = 8 waves; tile 64(M) x 128(N); waves 4(M) x 2(N), 4 wmma/K-step.
// A fragment: lane L row M=L&15, K runs [hi*8,+8) and [16+hi*8,+8).
// B fragment: lane L col N=L&15, K runs [hi*8,+8) and [16+hi*8,+8).
// ---------------------------------------------------------------------------
template <bool OUT_F32>
__global__ void k_wmma_gemm(const bf16_t* __restrict__ A,
                            const bf16_t* __restrict__ Bt,
                            const float*  __restrict__ bias,   // [O] or nullptr
                            void* __restrict__ Outv,
                            int O, int C, int P,
                            long long strideB, long long strideOut)
{
    __shared__ bf16_t Als[64 * APITCH];    // [m][k]
    __shared__ bf16_t Bls[128 * BPITCH];   // [n][k]

    const int b    = blockIdx.z;
    const int po   = blockIdx.x * 128;
    const int mo   = blockIdx.y * 64;
    const int t    = threadIdx.x;
    const int lane = t & 31;
    const int wv   = t >> 5;
    const int wm   = wv >> 1;              // 0..3
    const int wn   = wv & 1;               // 0..1

    const bf16_t* Bb = Bt + (size_t)b * strideB;

    v8f acc[4] = {};

    // staging indices
    const int arow = t >> 2;               // 0..63
    const int acol = (t & 3) * 8;          // 0,8,16,24
    const int brow = t >> 1;               // 0..127
    const int bcol = (t & 1) * 16;         // 0 or 16

    const bf16_t* ap = A  + (size_t)(mo + arow) * C + acol;
    const bf16_t* bp = Bb + (size_t)(po + brow) * C + bcol;

    for (int kk = 0; kk < C; kk += 32) {
        // stage A 64x32 (straight 16B copy)
        *(bf16x8*)&Als[arow * APITCH + acol] = *(const bf16x8*)(ap + kk);
        // stage B 128x32 (straight 32B copy, already [n][k])
        *(bf16x8*)&Bls[brow * BPITCH + bcol]     = *(const bf16x8*)(bp + kk);
        *(bf16x8*)&Bls[brow * BPITCH + bcol + 8] = *(const bf16x8*)(bp + kk + 8);
        if (kk + 32 < C) {                 // prefetch next K-step tiles
            __builtin_prefetch(ap + kk + 32, 0, 1);
            __builtin_prefetch(bp + kk + 32, 0, 1);
        }
        __syncthreads();

        Frag16 af;
        {
            int m  = wm * 16 + (lane & 15);
            int kb = (lane >> 4) * 8;
            af.h8[0] = *(const bf16x8*)&Als[m * APITCH + kb];
            af.h8[1] = *(const bf16x8*)&Als[m * APITCH + 16 + kb];
        }
#pragma unroll
        for (int j = 0; j < 4; ++j) {
            Frag16 bfrag;
            int n  = wn * 64 + j * 16 + (lane & 15);
            int kb = (lane >> 4) * 8;
            bfrag.h8[0] = *(const bf16x8*)&Bls[n * BPITCH + kb];
            bfrag.h8[1] = *(const bf16x8*)&Bls[n * BPITCH + 16 + kb];
            acc[j] = __builtin_amdgcn_wmma_f32_16x16x32_bf16(
                false, af.v, false, bfrag.v, (short)0, acc[j], false, false);
        }
        __syncthreads();
    }

    // epilogue: C/D layout -> M = r + (lane>>4)*8, N = lane&15
    const int mloc  = wm * 16 + (lane >> 4) * 8;
    const int nbase = po + wn * 64 + (lane & 15);
#pragma unroll
    for (int j = 0; j < 4; ++j) {
#pragma unroll
        for (int r = 0; r < 8; ++r) {
            int   m = mo + mloc + r;
            float v = acc[j][r];
            if (bias) v += bias[m];
            size_t off = (size_t)b * strideOut + (size_t)m * P + nbase + j * 16;
            if (OUT_F32) ((float*)Outv)[off] = v;
            else         ((bf16_t*)Outv)[off] = f2bf(v);
        }
    }
}

// ---------------------------------------------------------------------------
// High-path attention. Reference reshape semantics => per (b, head, c') the
// attention pairs rows (h even, h odd) with the feature axis along w.
// Writes pixel-major hxp_t[b][p = c'*64 + h][ch = head*64 + w].
// grid (c'=64, head=4, b=16), 256 threads (8 lanes per pair, 32 pairs).
// ---------------------------------------------------------------------------
__global__ void k_high_attn(const bf16_t* __restrict__ qkv,
                            bf16_t* __restrict__ hxpt)
{
    __shared__ bf16_t qs[4096], ks[4096], vs[4096];
    __shared__ bf16_t hxs[64 * 66];        // [w][h] with pad

    const int c = blockIdx.x, head = blockIdx.y, b = blockIdx.z;
    const int t = threadIdx.x;

    const size_t base = ((size_t)b * 768 + head * 64 + c) * 4096;
    const bf16_t* qg = qkv + base;
    const bf16_t* kg = qkv + base + (size_t)256 * 4096;
    const bf16_t* vg = qkv + base + (size_t)512 * 4096;

#pragma unroll
    for (int j = 0; j < 2; ++j) {
        int ch = t + j * 256;              // 512 chunks of 8 halves
        *(bf16x8*)&qs[ch * 8] = *(const bf16x8*)(qg + ch * 8);
        *(bf16x8*)&ks[ch * 8] = *(const bf16x8*)(kg + ch * 8);
        *(bf16x8*)&vs[ch * 8] = *(const bf16x8*)(vg + ch * 8);
    }
    __syncthreads();

    const int p  = t >> 3;                 // pair 0..31 -> rows 2p, 2p+1
    const int w0 = (t & 7) * 8;
    const bf16_t* q0 = &qs[(2 * p) * 64]; const bf16_t* q1 = q0 + 64;
    const bf16_t* k0 = &ks[(2 * p) * 64]; const bf16_t* k1 = k0 + 64;
    const bf16_t* v0 = &vs[(2 * p) * 64]; const bf16_t* v1 = v0 + 64;

    float s00 = 0.f, s01 = 0.f, s10 = 0.f, s11 = 0.f;
#pragma unroll
    for (int i = 0; i < 8; ++i) {
        float a0 = bf2f(q0[w0 + i]), a1 = bf2f(q1[w0 + i]);
        float c0 = bf2f(k0[w0 + i]), c1 = bf2f(k1[w0 + i]);
        s00 += a0 * c0; s01 += a0 * c1; s10 += a1 * c0; s11 += a1 * c1;
    }
#pragma unroll
    for (int m = 1; m < 8; m <<= 1) {      // reduce across the 8 lanes of a pair
        s00 += __shfl_xor(s00, m, 8);
        s01 += __shfl_xor(s01, m, 8);
        s10 += __shfl_xor(s10, m, 8);
        s11 += __shfl_xor(s11, m, 8);
    }
    const float SC = 0.125f;               // HEAD_DIM^-0.5
    s00 *= SC; s01 *= SC; s10 *= SC; s11 *= SC;
    float m0 = fmaxf(s00, s01), m1 = fmaxf(s10, s11);
    float e00 = __expf(s00 - m0), e01 = __expf(s01 - m0);
    float e10 = __expf(s10 - m1), e11 = __expf(s11 - m1);
    float p00 = e00 / (e00 + e01), p01 = e01 / (e00 + e01);
    float p10 = e10 / (e10 + e11), p11 = e11 / (e10 + e11);

#pragma unroll
    for (int i = 0; i < 8; ++i) {
        float vv0 = bf2f(v0[w0 + i]), vv1 = bf2f(v1[w0 + i]);
        hxs[(w0 + i) * 66 + 2 * p]     = f2bf(p00 * vv0 + p01 * vv1);
        hxs[(w0 + i) * 66 + 2 * p + 1] = f2bf(p10 * vv0 + p11 * vv1);
    }
    __syncthreads();

    // pixel-major writeback: hxp_t[b][c*64 + h][head*64 + w]
    int h  = t >> 2;
    int wb = (t & 3) * 16;
    bf16x8 o0, o1;
#pragma unroll
    for (int j = 0; j < 8; ++j) {
        o0[j] = hxs[(wb + j) * 66 + h];
        o1[j] = hxs[(wb + 8 + j) * 66 + h];
    }
    bf16_t* outp = hxpt + ((size_t)b * 4096 + c * 64 + h) * 256 + head * 64 + wb;
    *(bf16x8*)(outp)     = o0;
    *(bf16x8*)(outp + 8) = o1;
}

// ---------------------------------------------------------------------------
// bilinear 64->32 downsample (align-corners), bf16 in/out, channel-major
// ---------------------------------------------------------------------------
__global__ void k_downsample(const bf16_t* __restrict__ lq,
                             bf16_t* __restrict__ lqds)
{
    int i  = blockIdx.x * 256 + threadIdx.x;   // 16*256*1024
    int ox = i & 31, oy = (i >> 5) & 31, bc = i >> 10;
    const float f = 63.0f / 31.0f;
    float py = oy * f, px = ox * f;
    int y0 = min((int)floorf(py), 63); float wy = py - (float)y0; int y1 = min(y0 + 1, 63);
    int x0 = min((int)floorf(px), 63); float wx = px - (float)x0; int x1 = min(x0 + 1, 63);
    const bf16_t* src = lq + (((size_t)bc) << 12);
    float v00 = bf2f(src[y0 * 64 + x0]), v01 = bf2f(src[y0 * 64 + x1]);
    float v10 = bf2f(src[y1 * 64 + x0]), v11 = bf2f(src[y1 * 64 + x1]);
    float val = (v00 * (1.f - wy) + v10 * wy) * (1.f - wx)
              + (v01 * (1.f - wy) + v11 * wy) * wx;
    lqds[(((size_t)bc) << 10) + oy * 32 + ox] = f2bf(val);
}

// ---------------------------------------------------------------------------
// Low-path attention: per (b,c) S = Q*K^T*scale (no softmax!), X = S*K.
// 32x32x32 matmuls, 4 waves per workgroup, one 16x16 WMMA tile per wave per GEMM.
// ---------------------------------------------------------------------------
__global__ void k_low_attn(const bf16_t* __restrict__ lqds,
                           const bf16_t* __restrict__ kv,
                           bf16_t* __restrict__ lx)
{
    __shared__ bf16_t Qs [32 * APITCH];    // [i][w]
    __shared__ bf16_t Ks [32 * APITCH];    // [j][w]
    __shared__ bf16_t KTs[32 * APITCH];    // [w][j]
    __shared__ bf16_t Ss [32 * APITCH];    // [i][j] (bf16 S)

    const int c = blockIdx.x, b = blockIdx.y;
    const int t = threadIdx.x;             // 128 threads = 4 waves
    const int lane = t & 31, wv = t >> 5;
    const size_t base = ((size_t)b * 256 + c) * 1024;

    {
        int row = t >> 2, col = (t & 3) * 8;
        bf16x8 qv = *(const bf16x8*)(lqds + base + row * 32 + col);
        *(bf16x8*)&Qs[row * APITCH + col] = qv;
        bf16x8 kvv = *(const bf16x8*)(kv + base + row * 32 + col);
        *(bf16x8*)&Ks[row * APITCH + col] = kvv;
#pragma unroll
        for (int j = 0; j < 8; ++j) KTs[(col + j) * APITCH + row] = kvv[j];
    }
    __syncthreads();

    const int mi = (wv >> 1) * 16, nj = (wv & 1) * 16;

    // GEMM1: S = Q @ K^T
    Frag16 a1, b1;
    {
        int m  = mi + (lane & 15);
        int kb = (lane >> 4) * 8;
        a1.h8[0] = *(const bf16x8*)&Qs[m * APITCH + kb];
        a1.h8[1] = *(const bf16x8*)&Qs[m * APITCH + 16 + kb];
        int n   = nj + (lane & 15);
        int kb2 = (lane >> 4) * 8;
        b1.h8[0] = *(const bf16x8*)&Ks[n * APITCH + kb2];
        b1.h8[1] = *(const bf16x8*)&Ks[n * APITCH + 16 + kb2];
    }
    v8f s = {};
    s = __builtin_amdgcn_wmma_f32_16x16x32_bf16(false, a1.v, false, b1.v,
                                                (short)0, s, false, false);
    {
        int mb = mi + (lane >> 4) * 8, n = nj + (lane & 15);
#pragma unroll
        for (int r = 0; r < 8; ++r)
            Ss[(mb + r) * APITCH + n] = f2bf(s[r] * 0.125f);
    }
    __syncthreads();

    // GEMM2: X = S @ K
    Frag16 a2, b2;
    {
        int m  = mi + (lane & 15);
        int kb = (lane >> 4) * 8;
        a2.h8[0] = *(const bf16x8*)&Ss[m * APITCH + kb];
        a2.h8[1] = *(const bf16x8*)&Ss[m * APITCH + 16 + kb];
        int n   = nj + (lane & 15);
        int kb2 = (lane >> 4) * 8;
        b2.h8[0] = *(const bf16x8*)&KTs[n * APITCH + kb2];
        b2.h8[1] = *(const bf16x8*)&KTs[n * APITCH + 16 + kb2];
    }
    v8f xacc = {};
    xacc = __builtin_amdgcn_wmma_f32_16x16x32_bf16(false, a2.v, false, b2.v,
                                                   (short)0, xacc, false, false);
    {
        int mb = mi + (lane >> 4) * 8, n = nj + (lane & 15);
#pragma unroll
        for (int r = 0; r < 8; ++r)
            lx[base + (size_t)(mb + r) * 32 + n] = f2bf(xacc[r]);
    }
}

// ---------------------------------------------------------------------------
// bilinear 32->64 upsample (align-corners) fused with the output-channel
// permutation, written pixel-major:
//   lxp_t[b][p = y*64+x][ch = head*64+wfeat] = lxu[b, head*64+y, x, wfeat]
// ---------------------------------------------------------------------------
__global__ void k_upsample_perm(const bf16_t* __restrict__ lx,
                                bf16_t* __restrict__ lxpt)
{
    int i  = blockIdx.x * 256 + threadIdx.x;   // 16 * 4096 * 256
    int ch = i & 255;
    int p  = (i >> 8) & 4095;
    int b  = i >> 20;
    int y  = p >> 6, x = p & 63;
    int head = ch >> 6, wfeat = ch & 63;

    const float f = 31.0f / 63.0f;
    float pr = x * f, pc = wfeat * f;
    int r0 = min((int)floorf(pr), 31); float wr = pr - (float)r0; int r1 = min(r0 + 1, 31);
    int c0 = min((int)floorf(pc), 31); float wc = pc - (float)c0; int c1 = min(c0 + 1, 31);

    const bf16_t* src = lx + (((size_t)(b * 256 + head * 64 + y)) << 10);
    float v00 = bf2f(src[r0 * 32 + c0]), v01 = bf2f(src[r0 * 32 + c1]);
    float v10 = bf2f(src[r1 * 32 + c0]), v11 = bf2f(src[r1 * 32 + c1]);
    float val = (v00 * (1.f - wr) + v10 * wr) * (1.f - wc)
              + (v01 * (1.f - wr) + v11 * wr) * wc;
    lxpt[(((size_t)b) << 20) + ((size_t)p << 8) + ch] = f2bf(val);
}

// ---------------------------------------------------------------------------
extern "C" void kernel_launch(void* const* d_in, const int* in_sizes, int n_in,
                              void* d_out, int out_size, void* d_ws, size_t ws_size,
                              hipStream_t stream)
{
    (void)in_sizes; (void)n_in; (void)out_size; (void)ws_size;

    const float* x_f     = (const float*)d_in[0];
    const float* l_q_w   = (const float*)d_in[1];
    const float* l_kv_w  = (const float*)d_in[2];
    const float* l_p_w   = (const float*)d_in[3];
    const float* l_p_b   = (const float*)d_in[4];
    const float* h_qkv_w = (const float*)d_in[5];
    const float* h_p_w   = (const float*)d_in[6];
    const float* h_p_b   = (const float*)d_in[7];
    float* out = (float*)d_out;

    // workspace carve
    char* wsp = (char*)d_ws;
    auto carve = [&](size_t bytes) {
        void* p = (void*)wsp;
        wsp += (bytes + 255) & ~(size_t)255;
        return p;
    };
    bf16_t* Wqkv = (bf16_t*)carve((size_t)768 * 512 * 2);
    bf16_t* Wlq  = (bf16_t*)carve((size_t)256 * 512 * 2);
    bf16_t* Wkv  = (bf16_t*)carve((size_t)256 * 512 * 2);
    bf16_t* Whp  = (bf16_t*)carve((size_t)256 * 256 * 2);
    bf16_t* Wlp  = (bf16_t*)carve((size_t)256 * 256 * 2);
    bf16_t* xt   = (bf16_t*)carve((size_t)16 * 4096 * 512 * 2);  // [b][p][c]
    bf16_t* xpt  = (bf16_t*)carve((size_t)16 * 1024 * 512 * 2);  // [b][p'][c]
    bf16_t* qkvb = (bf16_t*)carve((size_t)16 * 768 * 4096 * 2);  // [b][o][p]
    bf16_t* hxpt = (bf16_t*)carve((size_t)16 * 4096 * 256 * 2);  // [b][p][ch]
    bf16_t* lqb  = (bf16_t*)carve((size_t)16 * 256 * 4096 * 2);  // [b][o][p]
    bf16_t* lqds = (bf16_t*)carve((size_t)16 * 256 * 1024 * 2);
    bf16_t* kvb  = (bf16_t*)carve((size_t)16 * 256 * 1024 * 2);
    bf16_t* lxb  = (bf16_t*)carve((size_t)16 * 256 * 1024 * 2);
    bf16_t* lxpt = (bf16_t*)carve((size_t)16 * 4096 * 256 * 2);  // [b][p][ch]

    // weight packs
    k_f32_to_bf16<<<768 * 512 / 256, 256, 0, stream>>>(h_qkv_w, Wqkv, 768 * 512);
    k_f32_to_bf16<<<256 * 512 / 256, 256, 0, stream>>>(l_q_w,  Wlq, 256 * 512);
    k_f32_to_bf16<<<256 * 512 / 256, 256, 0, stream>>>(l_kv_w, Wkv, 256 * 512);
    k_f32_to_bf16<<<256 * 256 / 256, 256, 0, stream>>>(h_p_w,  Whp, 256 * 256);
    k_f32_to_bf16<<<256 * 256 / 256, 256, 0, stream>>>(l_p_w,  Wlp, 256 * 256);

    // x -> pixel-major bf16, then 2x2 avg pool
    k_transpose_x<<<dim3(64, 16, 16), 256, 0, stream>>>(x_f, xt);
    k_pool_t<<<dim3(1024, 16), 256, 0, stream>>>(xt, xpt);

    // ---- high path ----
    k_wmma_gemm<false><<<dim3(32, 12, 16), 256, 0, stream>>>(
        Wqkv, xt, nullptr, (void*)qkvb, 768, 512, 4096,
        (long long)4096 * 512, (long long)768 * 4096);
    k_high_attn<<<dim3(64, 4, 16), 256, 0, stream>>>(qkvb, hxpt);
    k_wmma_gemm<true><<<dim3(32, 4, 16), 256, 0, stream>>>(
        Whp, hxpt, h_p_b, (void*)out, 256, 256, 4096,
        (long long)4096 * 256, (long long)512 * 4096);

    // ---- low path ----
    k_wmma_gemm<false><<<dim3(32, 4, 16), 256, 0, stream>>>(
        Wlq, xt, nullptr, (void*)lqb, 256, 512, 4096,
        (long long)4096 * 512, (long long)256 * 4096);
    k_downsample<<<16384, 256, 0, stream>>>(lqb, lqds);
    k_wmma_gemm<false><<<dim3(8, 4, 16), 256, 0, stream>>>(
        Wkv, xpt, nullptr, (void*)kvb, 256, 512, 1024,
        (long long)1024 * 512, (long long)256 * 1024);
    k_low_attn<<<dim3(256, 16), 128, 0, stream>>>(lqds, kvb, lxb);
    k_upsample_perm<<<65536, 256, 0, stream>>>(lxb, lxpt);
    k_wmma_gemm<true><<<dim3(32, 4, 16), 256, 0, stream>>>(
        Wlp, lxpt, l_p_b, (void*)(out + (size_t)256 * 4096), 256, 256, 4096,
        (long long)4096 * 256, (long long)512 * 4096);
}
